// ATT_Block_72078141161549
// MI455X (gfx1250) — compile-verified, matrix operand order checked by reading, not run on hardware
//
#include <hip/hip_runtime.h>
#include <hip/hip_bf16.h>
#include <math.h>

// ---------------------------------------------------------------------------
// Swin-style window-attention block for MI455X (gfx1250), wave32 + WMMA f16.
// Pipeline (all f16 intermediates, f32 accumulate in v_wmma_f32_16x16x32_f16):
//   ln1+window -> xw | qkv gemm (gather fused, section-specialized) -> Q,K,Vt
//   | attention -> O | proj gemm -> Y(f32) | ln2 + mlp -> out
// Workspace use: ~690 MB (5 x 96MB f16 tensors + 192MB f32 Y + weights).
// ---------------------------------------------------------------------------

typedef _Float16 f16;
typedef __attribute__((ext_vector_type(8)))  _Float16 v8h;
typedef __attribute__((ext_vector_type(16))) _Float16 v16h;
typedef __attribute__((ext_vector_type(8)))  float    v8f;

#define C_DIM   768
#define HEADS   12
#define HDIM    64
#define NWIN    1024
#define NTOK    64
#define KSTEP   32
#define S_THIRD 16777216u       /* 1024*64*256 = 2^24 elements */
#define S_MASK  (S_THIRD - 1u)

__device__ __forceinline__ v16h mk16(v8h lo, v8h hi) {
  v16h r;
#pragma unroll
  for (int i = 0; i < 8; ++i) { r[i] = lo[i]; r[i + 8] = hi[i]; }
  return r;
}

// 16-bit WMMA A/B fragment loader, row-major source, leading dim `ld` halfs.
// lane&15 selects the row (A) / column (B); per-lane K slice is two 8-half
// chunks: [k0 + h*8, +8) -> VGPR0-3 and [k0 + 16 + h*8, +8) -> VGPR4-7,
// h = lane>>4 (per CDNA5 ISA 7.12.2 16-bit A layout; B assumed mirrored).
__device__ __forceinline__ v16h frag_ld(const f16* p0, int ld, int k0, int lane) {
  int m = lane & 15, h = lane >> 4;
  const f16* p = p0 + (size_t)m * ld + k0 + h * 8;
  v8h lo = *(const v8h*)p;
  v8h hi = *(const v8h*)(p + 16);
  return mk16(lo, hi);
}

// ---------------- weight convert ----------------
__global__ void f32_to_f16_kernel(const float* __restrict__ src,
                                  f16* __restrict__ dst, int n) {
  int i = blockIdx.x * blockDim.x + threadIdx.x;
  if (i < n) dst[i] = (f16)src[i];
}

// ---------------- LN1 + window partition ----------------
__global__ __launch_bounds__(256) void ln1_window_kernel(
    const float* __restrict__ x, const float* __restrict__ g,
    const float* __restrict__ b, f16* __restrict__ xw) {
  int wave = threadIdx.x >> 5, lane = threadIdx.x & 31;
  int t = blockIdx.x * 8 + wave;               // token 0..65535
  const float* row = x + (size_t)t * C_DIM;
  float v[24];
  float s = 0.f, s2 = 0.f;
#pragma unroll
  for (int i = 0; i < 24; ++i) {
    float val = row[lane + i * 32];
    v[i] = val; s += val; s2 += val * val;
  }
#pragma unroll
  for (int m = 16; m; m >>= 1) { s += __shfl_xor(s, m, 32); s2 += __shfl_xor(s2, m, 32); }
  float mean = s * (1.f / C_DIM);
  float var  = s2 * (1.f / C_DIM) - mean * mean;
  float rinv = rsqrtf(var + 1e-5f);
  // (B,128,128) -> windows of 8x8
  int Bi = t >> 14, rem = t & 16383, hr = rem >> 7, wc = rem & 127;
  int win = (Bi * 16 + (hr >> 3)) * 16 + (wc >> 3);
  int n = (hr & 7) * 8 + (wc & 7);
  f16* dst = xw + (size_t)(win * 64 + n) * C_DIM;
#pragma unroll
  for (int i = 0; i < 24; ++i) {
    int c = lane + i * 32;
    dst[c] = (f16)((v[i] - mean) * rinv * g[c] + b[c]);
  }
}

// ---------------- QKV GEMM with fused red/green/blue gather ----------------
// Section-specialized at compile time (SEC: 0=q, 1=k, 2=v) so the K-loop is
// branch-free (no EXEC save/restore, no phi copies around the WMMAs).
// grid (1024 windows, 6), block 256 (8 waves). Each wave: one 16-col tile of
// this section's 768 outputs, 4 M-tiles (64 tokens), K=768 in steps of 32.
template <int SEC>
__global__ __launch_bounds__(256) void qkv_gemm_kernel(
    const f16* __restrict__ xw, const f16* __restrict__ wqkv,
    f16* __restrict__ outQK, f16* __restrict__ Vt) {
  int lane = threadIdx.x & 31, wv = threadIdx.x >> 5, h = lane >> 4;
  int w = blockIdx.x;
  int cout0 = (blockIdx.y * 8 + wv) * 16;      // 0..752 within section
  const f16* Bp = wqkv + (size_t)(SEC * C_DIM + cout0) * C_DIM;

  v8f acc[4] = {};
  for (int ks = 0; ks < C_DIM; ks += KSTEP) {
    v16h bf = frag_ld(Bp, C_DIM, ks, lane);
    __builtin_prefetch(Bp + (size_t)(lane & 15) * C_DIM + ks + KSTEP, 0, 1);
#pragma unroll
    for (int mt = 0; mt < 4; ++mt) {
      // flat index into window-ordered xw; red/green/blue via mod 2^24.
      unsigned i0 = (unsigned)((w * 64 + mt * 16 + (lane & 15)) * C_DIM + ks + h * 8);
      unsigned j0 = i0 & S_MASK;
      unsigned j1 = (i0 + 16) & S_MASK;        // chunks never straddle the wrap
      v8h lo, hi;
      if constexpr (SEC == 0) {                // q input = green + blue
        lo = *(const v8h*)(xw + S_THIRD + j0) + *(const v8h*)(xw + 2u * S_THIRD + j0);
        hi = *(const v8h*)(xw + S_THIRD + j1) + *(const v8h*)(xw + 2u * S_THIRD + j1);
      } else {                                 // k/v input = red
        lo = *(const v8h*)(xw + j0);
        hi = *(const v8h*)(xw + j1);
      }
      v16h af = mk16(lo, hi);
      acc[mt] = __builtin_amdgcn_wmma_f32_16x16x32_f16(
          false, af, false, bf, (short)0, acc[mt], false, false);
    }
  }

  if constexpr (SEC < 2) {                     // Q / K: [win][tok][c] f16
    f16* outb = outQK + (size_t)w * NTOK * C_DIM + cout0 + (lane & 15);
#pragma unroll
    for (int mt = 0; mt < 4; ++mt)
#pragma unroll
      for (int r = 0; r < 8; ++r)
        outb[(size_t)(mt * 16 + r + 8 * h) * C_DIM] = (f16)acc[mt][r];
  } else {                                     // V stored transposed: [win][c][tok]
    f16* outb = Vt + (size_t)w * C_DIM * NTOK + (size_t)(cout0 + (lane & 15)) * NTOK;
#pragma unroll
    for (int mt = 0; mt < 4; ++mt) {
      v8h pk;
#pragma unroll
      for (int r = 0; r < 8; ++r) pk[r] = (f16)acc[mt][r];
      *(v8h*)(outb + mt * 16 + 8 * h) = pk;    // rows contiguous -> b128 store
    }
  }
}

// ---------------- attention: one wave per (window, head) ----------------
__global__ __launch_bounds__(64) void attn_kernel(
    const f16* __restrict__ Q, const f16* __restrict__ Kb,
    const f16* __restrict__ Vt, f16* __restrict__ O) {
  __shared__ f16 probs[2][NTOK * NTOK];        // 16 KB, per-wave scratch
  int lane = threadIdx.x & 31, wv = threadIdx.x >> 5, h = lane >> 4;
  int g = blockIdx.x * 2 + wv;
  int w = g / HEADS, hd = g % HEADS;
  const f16* Qh = Q  + (size_t)w * NTOK * C_DIM + hd * HDIM;
  const f16* Kh = Kb + (size_t)w * NTOK * C_DIM + hd * HDIM;
  const f16* Vh = Vt + (size_t)w * C_DIM * NTOK + (size_t)hd * HDIM * NTOK;

  // scores = q @ k^T  (64x64, 16 accum tiles, K=64)
  v8f s[4][4] = {};
#pragma unroll
  for (int ks = 0; ks < HDIM; ks += KSTEP) {
    v16h af[4], bf[4];
#pragma unroll
    for (int i = 0; i < 4; ++i) {
      af[i] = frag_ld(Qh + i * 16 * C_DIM, C_DIM, ks, lane);
      bf[i] = frag_ld(Kh + i * 16 * C_DIM, C_DIM, ks, lane);
    }
#pragma unroll
    for (int mt = 0; mt < 4; ++mt)
#pragma unroll
      for (int nt = 0; nt < 4; ++nt)
        s[mt][nt] = __builtin_amdgcn_wmma_f32_16x16x32_f16(
            false, af[mt], false, bf[nt], (short)0, s[mt][nt], false, false);
  }

  // softmax over rows (row = vgpr slot, cols across 16-lane half-groups)
  const float scale = 0.125f;                  // D^-0.5, D=64
#pragma unroll
  for (int mt = 0; mt < 4; ++mt)
#pragma unroll
    for (int r = 0; r < 8; ++r) {
      float mval = -3.0e38f;
#pragma unroll
      for (int nt = 0; nt < 4; ++nt) mval = fmaxf(mval, s[mt][nt][r]);
#pragma unroll
      for (int m = 8; m; m >>= 1) mval = fmaxf(mval, __shfl_xor(mval, m, 32));
      float sum = 0.f;
      float e[4];
#pragma unroll
      for (int nt = 0; nt < 4; ++nt) {
        e[nt] = __expf(scale * (s[mt][nt][r] - mval));
        sum += e[nt];
      }
#pragma unroll
      for (int m = 8; m; m >>= 1) sum += __shfl_xor(sum, m, 32);
      float rs = 1.f / sum;
      int row = mt * 16 + r + 8 * h;
#pragma unroll
      for (int nt = 0; nt < 4; ++nt)
        probs[wv][row * NTOK + nt * 16 + (lane & 15)] = (f16)(e[nt] * rs);
    }

  // out = P @ V  (A from LDS re-layout, B from transposed V -> contiguous K)
  v8f o[4][4] = {};
#pragma unroll
  for (int ks = 0; ks < NTOK; ks += KSTEP) {
    v16h af[4], bf[4];
#pragma unroll
    for (int i = 0; i < 4; ++i) {
      af[i] = frag_ld(&probs[wv][i * 16 * NTOK], NTOK, ks, lane);
      bf[i] = frag_ld(Vh + i * 16 * NTOK, NTOK, ks, lane);
    }
#pragma unroll
    for (int mt = 0; mt < 4; ++mt)
#pragma unroll
      for (int nt = 0; nt < 4; ++nt)
        o[mt][nt] = __builtin_amdgcn_wmma_f32_16x16x32_f16(
            false, af[mt], false, bf[nt], (short)0, o[mt][nt], false, false);
  }

  f16* Ob = O + (size_t)w * NTOK * C_DIM + hd * HDIM;
#pragma unroll
  for (int mt = 0; mt < 4; ++mt)
#pragma unroll
    for (int nt = 0; nt < 4; ++nt)
#pragma unroll
      for (int r = 0; r < 8; ++r)
        Ob[(size_t)(mt * 16 + r + 8 * h) * C_DIM + nt * 16 + (lane & 15)] =
            (f16)o[mt][nt][r];
}

// ---------------- output projection -> f32 Y ----------------
__global__ __launch_bounds__(256) void proj_gemm_kernel(
    const f16* __restrict__ O, const f16* __restrict__ Wp,
    const float* __restrict__ pb, float* __restrict__ Y) {
  int lane = threadIdx.x & 31, wv = threadIdx.x >> 5, h = lane >> 4;
  int w = blockIdx.x;
  int cout0 = (blockIdx.y * 8 + wv) * 16;
  const f16* Ap = O + (size_t)w * NTOK * C_DIM;
  const f16* Bp = Wp + (size_t)cout0 * C_DIM;
  v8f acc[4] = {};
  for (int ks = 0; ks < C_DIM; ks += KSTEP) {
    v16h bf = frag_ld(Bp, C_DIM, ks, lane);
    __builtin_prefetch(Bp + (size_t)(lane & 15) * C_DIM + ks + KSTEP, 0, 1);
#pragma unroll
    for (int mt = 0; mt < 4; ++mt) {
      v16h af = frag_ld(Ap + mt * 16 * C_DIM, C_DIM, ks, lane);
      acc[mt] = __builtin_amdgcn_wmma_f32_16x16x32_f16(
          false, af, false, bf, (short)0, acc[mt], false, false);
    }
  }
  float bias = pb[cout0 + (lane & 15)];
  float* Yb = Y + (size_t)w * NTOK * C_DIM + cout0 + (lane & 15);
#pragma unroll
  for (int mt = 0; mt < 4; ++mt)
#pragma unroll
    for (int r = 0; r < 8; ++r)
      Yb[(size_t)(mt * 16 + r + 8 * h) * C_DIM] = acc[mt][r] + bias;
}

// ---------------- LN2 + C->1->C MLP (exact GELU) ----------------
__global__ __launch_bounds__(256) void ln2_mlp_kernel(
    const float* __restrict__ Y, const float* __restrict__ g2,
    const float* __restrict__ b2, const float* __restrict__ fc1w,
    const float* __restrict__ fc1b, const float* __restrict__ fc2w,
    const float* __restrict__ fc2b, float* __restrict__ out) {
  int wv = threadIdx.x >> 5, lane = threadIdx.x & 31;
  int t = blockIdx.x * 8 + wv;
  const float* row = Y + (size_t)t * C_DIM;
  float v[24];
  float s = 0.f, s2 = 0.f;
#pragma unroll
  for (int i = 0; i < 24; ++i) {
    float val = row[lane + i * 32];
    v[i] = val; s += val; s2 += val * val;
  }
#pragma unroll
  for (int m = 16; m; m >>= 1) { s += __shfl_xor(s, m, 32); s2 += __shfl_xor(s2, m, 32); }
  float mean = s * (1.f / C_DIM);
  float var  = s2 * (1.f / C_DIM) - mean * mean;
  float rinv = rsqrtf(var + 1e-5f);
  float dot = 0.f;
#pragma unroll
  for (int i = 0; i < 24; ++i) {
    int c = lane + i * 32;
    float z = (v[i] - mean) * rinv * g2[c] + b2[c];
    dot += z * fc1w[c];
  }
#pragma unroll
  for (int m = 16; m; m >>= 1) dot += __shfl_xor(dot, m, 32);
  float u = dot + fc1b[0];
  float gel = 0.5f * u * (1.f + erff(u * 0.70710678118f));  // exact gelu
  float* orow = out + (size_t)t * C_DIM;
#pragma unroll
  for (int i = 0; i < 24; ++i) {
    int c = lane + i * 32;
    orow[c] = gel * fc2w[c] + fc2b[c];
  }
}

// ---------------------------------------------------------------------------
extern "C" void kernel_launch(void* const* d_in, const int* in_sizes, int n_in,
                              void* d_out, int out_size, void* d_ws, size_t ws_size,
                              hipStream_t stream) {
  (void)in_sizes; (void)n_in; (void)out_size; (void)ws_size;
  const float* x    = (const float*)d_in[0];
  const float* g1   = (const float*)d_in[1];
  const float* b1   = (const float*)d_in[2];
  const float* g2   = (const float*)d_in[3];
  const float* b2   = (const float*)d_in[4];
  const float* qkvw = (const float*)d_in[5];
  const float* pw   = (const float*)d_in[6];
  const float* pb   = (const float*)d_in[7];
  const float* f1w  = (const float*)d_in[8];
  const float* f1b  = (const float*)d_in[9];
  const float* f2w  = (const float*)d_in[10];
  const float* f2b  = (const float*)d_in[11];
  float* out = (float*)d_out;

  const size_t TOK_ELEMS = (size_t)NWIN * NTOK * C_DIM;   // 50,331,648
  char* ws = (char*)d_ws;
  size_t off = 0;
  auto take = [&](size_t bytes) -> char* {
    char* p = ws + off;
    off += (bytes + 255) & ~(size_t)255;
    return p;
  };
  f16*   qkvh  = (f16*)take((size_t)3 * C_DIM * C_DIM * sizeof(f16));
  f16*   projh = (f16*)take((size_t)C_DIM * C_DIM * sizeof(f16));
  f16*   xwh   = (f16*)take(TOK_ELEMS * sizeof(f16));
  f16*   Qb    = (f16*)take(TOK_ELEMS * sizeof(f16));
  f16*   Kbuf  = (f16*)take(TOK_ELEMS * sizeof(f16));
  f16*   Vtb   = (f16*)take(TOK_ELEMS * sizeof(f16));
  f16*   Ob    = (f16*)take(TOK_ELEMS * sizeof(f16));
  float* Yb    = (float*)take(TOK_ELEMS * sizeof(float));

  int nqkv = 3 * C_DIM * C_DIM, nproj = C_DIM * C_DIM;
  f32_to_f16_kernel<<<(nqkv + 255) / 256, 256, 0, stream>>>(qkvw, qkvh, nqkv);
  f32_to_f16_kernel<<<(nproj + 255) / 256, 256, 0, stream>>>(pw, projh, nproj);

  ln1_window_kernel<<<65536 / 8, 256, 0, stream>>>(x, g1, b1, xwh);

  qkv_gemm_kernel<0><<<dim3(NWIN, 6), 256, 0, stream>>>(xwh, qkvh, Qb,   Vtb);
  qkv_gemm_kernel<1><<<dim3(NWIN, 6), 256, 0, stream>>>(xwh, qkvh, Kbuf, Vtb);
  qkv_gemm_kernel<2><<<dim3(NWIN, 6), 256, 0, stream>>>(xwh, qkvh, Qb,   Vtb);

  attn_kernel<<<(NWIN * HEADS) / 2, 64, 0, stream>>>(Qb, Kbuf, Vtb, Ob);

  proj_gemm_kernel<<<dim3(NWIN, 6), 256, 0, stream>>>(Ob, projh, pb, Yb);

  ln2_mlp_kernel<<<65536 / 8, 256, 0, stream>>>(Yb, g2, b2, f1w, f1b, f2w, f2b, out);
}
